// MLGO_10831907520966
// MI455X (gfx1250) — matrix operand order repaired and checked by reading, not compiled, not run on hardware
//
#include <hip/hip_runtime.h>

// ---------------------------------------------------------------------------
// MLGO forward for MI455X (gfx1250, wave32).
// Heavy ops (isip graph GEMMs ~206 GFLOP, tconv MLP ~29 GFLOP) run on the
// matrix cores via v_wmma_f32_16x16x32_bf16 (f32 accumulate). Master state is
// fp32; a bf16 transposed copy Zt[f][n] is maintained so A/B WMMA fragments
// are plain contiguous b128 global loads (both operands use k = 16*half + s).
// Everything is L2-resident (<< 192MB), so the win is matrix-op throughput.
// ---------------------------------------------------------------------------

typedef __attribute__((ext_vector_type(16))) __bf16 v16bf;
typedef __attribute__((ext_vector_type(8)))  __bf16 v8bf;
typedef __attribute__((ext_vector_type(8)))  float  v8f;

#define Bdim 4
#define Ndim 1024
#define Tdim 12
#define Cdim 64
#define Ddim 40
#define Fdim 3072           // B*C*T, f = b*768 + c*12 + t
#define SZ   3145728        // Ndim*Fdim
#define NN   1048576        // Ndim*Ndim

// ===========================================================================
// isip GEMM step: zout[n,f] = zin[n,f] + c*( (Ahat @ zin)[n,f] - zin[n,f] )
// Ahat: bf16 [N,N] row-major.  ztin: bf16 [F,N] (= zin transposed).
// Each wave computes a 16(M) x 128(F) strip; 8 waves stack in M (128x128/WG).
// ===========================================================================
__global__ __launch_bounds__(256) void k_isip(
    const __bf16* __restrict__ Ahat,
    const float*  __restrict__ zin,  const __bf16* __restrict__ ztin,
    float*        __restrict__ zout, __bf16*       __restrict__ ztout,
    const float*  __restrict__ alpha, float step)
{
  const float cf = step * alpha[0];
  const int wave = threadIdx.x >> 5;
  const int l = threadIdx.x & 31, hf = l >> 4, lc = l & 15;
  const int mbase = blockIdx.y * 128 + wave * 16;
  const int fbase = blockIdx.x * 128;

  v8f acc[8] = {};
  for (int k = 0; k < Ndim; k += 32) {
    const __bf16* pa = Ahat + (size_t)(mbase + lc) * Ndim + k + 16 * hf;
    v8bf alo = *(const v8bf*)pa;
    v8bf ahi = *(const v8bf*)(pa + 8);
    v16bf af = __builtin_shufflevector(alo, ahi,
        0,1,2,3,4,5,6,7,8,9,10,11,12,13,14,15);
#pragma unroll
    for (int j = 0; j < 8; ++j) {
      const __bf16* pb = ztin + (size_t)(fbase + j*16 + lc) * Ndim + k + 16 * hf;
      v16bf bf = *(const v16bf*)pb;
      acc[j] = __builtin_amdgcn_wmma_f32_16x16x32_bf16(
          false, af, false, bf, (short)0, acc[j], false, false);
    }
  }
  const int mrow0 = mbase + 8 * hf;
#pragma unroll
  for (int j = 0; j < 8; ++j) {
    const int f = fbase + j*16 + lc;
    v8bf tb;
#pragma unroll
    for (int v = 0; v < 8; ++v) {
      const int m = mrow0 + v;
      float zo = zin[(size_t)m * Fdim + f];
      float val = zo + cf * (acc[j][v] - zo);
      zout[(size_t)m * Fdim + f] = val;
      tb[v] = (__bf16)val;
    }
    *(v8bf*)(ztout + (size_t)f * Ndim + mrow0) = tb;
  }
}

// ===========================================================================
// tconv: per-row MLP over T:  12 -> 64 -> 64 x3 -> 12, all via WMMA.
// Rows are (n,b,c) triples; 16 rows per wave; weights pre-transposed/padded
// to bf16; inter-layer activations ping-pong through per-wave LDS tiles.
// ===========================================================================
__global__ __launch_bounds__(256) void k_tconv(
    const float* __restrict__ zin, float* __restrict__ zout,
    __bf16* __restrict__ ztout,
    const __bf16* __restrict__ wt_in, const __bf16* __restrict__ wt_h,
    const __bf16* __restrict__ wt_out,
    const float* __restrict__ b_in, const float* __restrict__ b_h,
    const float* __restrict__ b_out)
{
  __shared__ __bf16 lds[8][2][16 * 64];
  const int wave = threadIdx.x >> 5;
  const int l = threadIdx.x & 31, hf = l >> 4, lc = l & 15;
  const int wt = blockIdx.x * 8 + wave;     // 0..16383
  const int n = wt >> 4;
  const int r2base = (wt & 15) * 16;        // (b,c) row base

  // ---- load 16 fp32 rows of length 12, build padded bf16 A fragment -------
  const float* rowp = zin + (size_t)n * Fdim + (size_t)(r2base + lc) * Tdim;
  float rv[12];
#pragma unroll
  for (int i = 0; i < 3; ++i) {
    float4 q = ((const float4*)rowp)[i];
    rv[4*i] = q.x; rv[4*i+1] = q.y; rv[4*i+2] = q.z; rv[4*i+3] = q.w;
  }
  v16bf a0;
#pragma unroll
  for (int i = 0; i < 8; ++i) {
    int t = 16*hf + i;      a0[i]   = (t < 12) ? (__bf16)rv[t] : (__bf16)0.0f;
  }
#pragma unroll
  for (int i = 0; i < 8; ++i) {
    int t = 16*hf + 8 + i;  a0[8+i] = (t < 12) ? (__bf16)rv[t] : (__bf16)0.0f;
  }

  // ---- input layer: [16x32(pad12)] x [32x64] ------------------------------
  v8f acc[4];
  const v8f zf = {};
#pragma unroll
  for (int j = 0; j < 4; ++j) {
    v16bf bf = *(const v16bf*)(wt_in + (j*16 + lc) * 32 + 16 * hf);
    acc[j] = __builtin_amdgcn_wmma_f32_16x16x32_bf16(
        false, a0, false, bf, (short)0, zf, false, false);
  }
  __bf16* cur = lds[wave][0];
#pragma unroll
  for (int j = 0; j < 4; ++j)
#pragma unroll
    for (int v = 0; v < 8; ++v) {
      float x = acc[j][v] + b_in[j*16 + lc];
      cur[(v + 8*hf) * 64 + j*16 + lc] = (__bf16)(x > 0.f ? x : 0.f);
    }

  // ---- 3 hidden layers: [16x64] x [64x64] ---------------------------------
  for (int layer = 0; layer < 3; ++layer) {
    const __bf16* W = wt_h + layer * 4096;
    const float* bb = b_h + layer * 64;
    __bf16* nxt = lds[wave][(layer + 1) & 1];
    v8f ha[4] = {};
#pragma unroll
    for (int kc = 0; kc < 2; ++kc) {
      const __bf16* pa = cur + lc * 64 + kc * 32 + 16 * hf;
      v8bf alo = *(const v8bf*)pa;
      v8bf ahi = *(const v8bf*)(pa + 8);
      v16bf af = __builtin_shufflevector(alo, ahi,
          0,1,2,3,4,5,6,7,8,9,10,11,12,13,14,15);
#pragma unroll
      for (int j = 0; j < 4; ++j) {
        v16bf bf = *(const v16bf*)(W + (j*16 + lc) * 64 + kc * 32 + 16 * hf);
        ha[j] = __builtin_amdgcn_wmma_f32_16x16x32_bf16(
            false, af, false, bf, (short)0, ha[j], false, false);
      }
    }
#pragma unroll
    for (int j = 0; j < 4; ++j)
#pragma unroll
      for (int v = 0; v < 8; ++v) {
        float x = ha[j][v] + bb[j*16 + lc];
        nxt[(v + 8*hf) * 64 + j*16 + lc] = (__bf16)(x > 0.f ? x : 0.f);
      }
    cur = nxt;
  }

  // ---- output layer: [16x64] x [64x16(pad12)], no relu --------------------
  v8f oa = {};
#pragma unroll
  for (int kc = 0; kc < 2; ++kc) {
    const __bf16* pa = cur + lc * 64 + kc * 32 + 16 * hf;
    v8bf alo = *(const v8bf*)pa;
    v8bf ahi = *(const v8bf*)(pa + 8);
    v16bf af = __builtin_shufflevector(alo, ahi,
        0,1,2,3,4,5,6,7,8,9,10,11,12,13,14,15);
    v16bf bf = *(const v16bf*)(wt_out + lc * 64 + kc * 32 + 16 * hf);
    oa = __builtin_amdgcn_wmma_f32_16x16x32_bf16(
        false, af, false, bf, (short)0, oa, false, false);
  }
  if (lc < 12) {
    float bo = b_out[lc];
#pragma unroll
    for (int v = 0; v < 8; ++v) {
      int r = r2base + v + 8*hf;
      float x = oa[v] + bo;
      zout[(size_t)n * Fdim + (size_t)r * Tdim + lc] = x;
      ztout[((size_t)r * Tdim + lc) * Ndim + n] = (__bf16)x;
    }
  }
}

// ===========================================================================
// VALU kernels (small-FLOP glue)
// ===========================================================================
__global__ void k_start(const float* __restrict__ in,
                        const float* __restrict__ sw, const float* __restrict__ sb,
                        float* __restrict__ s0, float* __restrict__ s1,
                        float* __restrict__ s2)
{
  int idx = blockIdx.x * 256 + threadIdx.x;
  int t = idx % 12, c = (idx / 12) & 63, b = (idx / 768) & 3, n = idx / 3072;
  float v = in[((size_t)b * Ndim + n) * Tdim + t] * sw[c] + sb[c];
  s0[idx] = v; s1[idx] = v; s2[idx] = v;
}

__global__ void k_conv_cat_relu(const float* __restrict__ X1,
                                const float* __restrict__ X2,
                                const float* __restrict__ W,
                                const float* __restrict__ bias,
                                float* __restrict__ out, __bf16* __restrict__ zt)
{
  int idx = blockIdx.x * 256 + threadIdx.x;
  int t = idx % 12, o = (idx / 12) & 63, b = (idx / 768) & 3, n = idx / 3072;
  const float* p1 = X1 + (size_t)n * Fdim + b * 768 + t;
  const float* p2 = X2 + (size_t)n * Fdim + b * 768 + t;
  const float* w = W + o * 128;
  float s = bias[o];
#pragma unroll 8
  for (int c = 0; c < 64; ++c) s += w[c] * p1[c * 12];
#pragma unroll 8
  for (int c = 0; c < 64; ++c) s += w[64 + c] * p2[c * 12];
  s = s > 0.f ? s : 0.f;
  out[idx] = s;
  if (zt) zt[((size_t)(b * 64 + o) * 12 + t) * Ndim + n] = (__bf16)s;
}

__global__ void k_axpy(float* __restrict__ s, const float* __restrict__ d)
{
  int idx = blockIdx.x * 256 + threadIdx.x;
  s[idx] += 0.25f * d[idx];
}

__global__ void k_addinto(float* __restrict__ d, const float* __restrict__ a)
{
  int idx = blockIdx.x * 256 + threadIdx.x;
  d[idx] += a[idx];
}

__global__ void k_embmlp(const float* __restrict__ emb, const float* __restrict__ W,
                         const float* __restrict__ b, float* __restrict__ out)
{
  int idx = blockIdx.x * 256 + threadIdx.x;          // N*D = 40960
  int n = idx / 40, j = idx % 40;
  float s = b[j];
  for (int d = 0; d < 40; ++d) s += emb[n * 40 + d] * W[j * 40 + d];
  out[idx] = tanhf(3.0f * s);
}

__global__ void k_adjraw(const float* __restrict__ n1, const float* __restrict__ n2,
                         float* __restrict__ out)
{
  int idx = blockIdx.x * 256 + threadIdx.x;          // N*N
  int n = idx >> 10, m = idx & 1023;
  float s1 = 0.f, s2 = 0.f;
  for (int d = 0; d < 40; ++d) {
    s1 += n1[n * 40 + d] * n2[m * 40 + d];
    s2 += n2[n * 40 + d] * n1[m * 40 + d];
  }
  float a = tanhf(3.0f * (s1 - s2));
  out[idx] = a > 0.f ? a : 0.f;
}

__global__ __launch_bounds__(256) void k_topk(const float* __restrict__ a,
                                              float* __restrict__ out)
{
  __shared__ float vals[1024];
  __shared__ float rv[256];
  __shared__ int   ri[256];
  int row = blockIdx.x, tid = threadIdx.x;
  for (int m = tid; m < 1024; m += 256) {
    vals[m] = a[row * 1024 + m];
    out[row * 1024 + m] = 0.f;
  }
  __syncthreads();
  for (int it = 0; it < 20; ++it) {
    float bv = -1e30f; int bi = 1 << 30;
    for (int m = tid; m < 1024; m += 256) {
      float v = vals[m];
      if (v > bv || (v == bv && m < bi)) { bv = v; bi = m; }
    }
    rv[tid] = bv; ri[tid] = bi;
    __syncthreads();
    for (int s = 128; s > 0; s >>= 1) {
      if (tid < s) {
        float ov = rv[tid + s]; int oi = ri[tid + s];
        if (ov > rv[tid] || (ov == rv[tid] && oi < ri[tid])) { rv[tid] = ov; ri[tid] = oi; }
      }
      __syncthreads();
    }
    if (tid == 0) {
      int b0 = ri[0];
      out[row * 1024 + b0] = a[row * 1024 + b0];
      vals[b0] = -1e30f;
    }
    __syncthreads();
  }
}

// Ahat = rownorm(A + I) cast to bf16; transposed=1 uses A^T.
__global__ __launch_bounds__(256) void k_rownorm(const float* __restrict__ A,
                                                 __bf16* __restrict__ out,
                                                 int transposed)
{
  __shared__ float red[256];
  int row = blockIdx.x, tid = threadIdx.x;
  float part = 0.f;
  for (int m = tid; m < 1024; m += 256) {
    float v = transposed ? A[(size_t)m * 1024 + row] : A[(size_t)row * 1024 + m];
    if (m == row) v += 1.f;
    part += v;
  }
  red[tid] = part; __syncthreads();
  for (int s = 128; s > 0; s >>= 1) {
    if (tid < s) red[tid] += red[tid + s];
    __syncthreads();
  }
  float inv = 1.f / red[0];
  for (int m = tid; m < 1024; m += 256) {
    float v = transposed ? A[(size_t)m * 1024 + row] : A[(size_t)row * 1024 + m];
    if (m == row) v += 1.f;
    out[(size_t)row * 1024 + m] = (__bf16)(v * inv);
  }
}

// Transpose/pad tconv weights to bf16, K-contiguous for WMMA B operands.
__global__ void k_prep_tcw(const float* __restrict__ tin, const float* __restrict__ th,
                           const float* __restrict__ tout,
                           __bf16* __restrict__ wt_in, __bf16* __restrict__ wt_h,
                           __bf16* __restrict__ wt_out)
{
  int tid = threadIdx.x;
  for (int idx = tid; idx < 64 * 32; idx += 256) {           // wt_in[o][k]
    int o = idx / 32, k = idx % 32;
    wt_in[idx] = (k < 12) ? (__bf16)tin[k * 64 + o] : (__bf16)0.0f;
  }
  for (int idx = tid; idx < 3 * 64 * 64; idx += 256) {       // wt_h[l][g][h]
    int lyr = idx / 4096, r = idx % 4096, g = r / 64, hh = r % 64;
    wt_h[idx] = (__bf16)th[lyr * 4096 + hh * 64 + g];
  }
  for (int idx = tid; idx < 16 * 64; idx += 256) {           // wt_out[t][h]
    int t = idx / 64, hh = idx % 64;
    wt_out[idx] = (t < 12) ? (__bf16)tout[hh * 12 + t] : (__bf16)0.0f;
  }
}

__global__ void k_lnstat(const float* __restrict__ x, float* __restrict__ stats)
{
  __shared__ float sv[256], sq[256];
  int n = blockIdx.x, b = blockIdx.y, tid = threadIdx.x;
  const float* p = x + (size_t)n * Fdim + b * 768;
  float s = 0.f, q = 0.f;
  for (int i = tid; i < 768; i += 256) { float v = p[i]; s += v; q += v * v; }
  sv[tid] = s; sq[tid] = q; __syncthreads();
  for (int st = 128; st > 0; st >>= 1) {
    if (tid < st) { sv[tid] += sv[tid + st]; sq[tid] += sq[tid + st]; }
    __syncthreads();
  }
  if (tid == 0) {
    atomicAdd(&stats[b * 2], sv[0]);
    atomicAdd(&stats[b * 2 + 1], sq[0]);
  }
}

__global__ void k_lnfin(float* __restrict__ stats)
{
  int tid = threadIdx.x;
  if (tid < 4) {
    const float CNT = 786432.0f;  // C*N*T
    float mean = stats[tid * 2] / CNT;
    float var = stats[tid * 2 + 1] / CNT - mean * mean;
    stats[8 + tid] = mean;
    stats[12 + tid] = rsqrtf(var + 1e-5f);
  }
}

__global__ void k_head0(const float* __restrict__ x, const float* __restrict__ stats,
                        const float* __restrict__ aw, const float* __restrict__ ab,
                        const float* __restrict__ W, const float* __restrict__ bias,
                        float* __restrict__ out)
{
  int idx = blockIdx.x * 256 + threadIdx.x;
  int t = idx % 12, o = (idx / 12) & 63, b = (idx / 768) & 3, n = idx / 3072;
  float mu = stats[8 + b], rstd = stats[12 + b];
  const float* xp = x + (size_t)n * Fdim + b * 768 + t;
  float s = bias[o];
  for (int c = 0; c < 64; ++c) {
    float xn = (xp[c * 12] - mu) * rstd * aw[c * 1024 + n] + ab[c * 1024 + n];
    s += W[o * 64 + c] * xn;
  }
  out[idx] = s > 0.f ? s : 0.f;
}

__global__ void k_head1(const float* __restrict__ e0, const float* __restrict__ W,
                        const float* __restrict__ bias, float* __restrict__ out)
{
  int idx = blockIdx.x * 256 + threadIdx.x;          // N*B*32*T
  int t = idx % 12, o = (idx / 12) & 31, b = (idx / 384) & 3, n = idx / 1536;
  const float* p = e0 + (size_t)n * Fdim + b * 768 + t;
  float s = bias[o];
  for (int c = 0; c < 64; ++c) s += W[o * 64 + c] * p[c * 12];
  out[idx] = s > 0.f ? s : 0.f;
}

__global__ void k_head2(const float* __restrict__ e1, const float* __restrict__ W,
                        const float* __restrict__ bias, float* __restrict__ out)
{
  int idx = blockIdx.x * 256 + threadIdx.x;          // N*B*T
  int t = idx % 12, b = (idx / 12) & 3, n = idx / 48;
  const float* p = e1 + (size_t)n * 1536 + b * 384 + t;
  float s = bias[0];
  for (int c = 0; c < 32; ++c) s += W[c] * p[c * 12];
  out[((size_t)b * Ndim + n) * Tdim + t] = s;
}

// ===========================================================================
// Host launch
// ===========================================================================
static inline char* carve(char*& p, size_t bytes) {
  char* r = p;
  p += (bytes + 255) & ~(size_t)255;
  return r;
}

extern "C" void kernel_launch(void* const* d_in, const int* in_sizes, int n_in,
                              void* d_out, int out_size, void* d_ws, size_t ws_size,
                              hipStream_t stream) {
  (void)in_sizes; (void)n_in; (void)out_size; (void)ws_size;
  const float* input  = (const float*)d_in[0];
  const float* adjA1  = (const float*)d_in[1];
  const float* adjA2  = (const float*)d_in[2];
  const float* start_w = (const float*)d_in[3];
  const float* start_b = (const float*)d_in[4];
  const float* mlp1_w = (const float*)d_in[5];
  const float* mlp1_b = (const float*)d_in[6];
  const float* mlp2_w = (const float*)d_in[7];
  const float* mlp2_b = (const float*)d_in[8];
  const float* mlp3_w = (const float*)d_in[9];
  const float* mlp3_b = (const float*)d_in[10];
  const float* tc_in_w = (const float*)d_in[11];
  const float* tc_in_b = (const float*)d_in[12];
  const float* tc_h_w  = (const float*)d_in[13];
  const float* tc_h_b  = (const float*)d_in[14];
  const float* tc_out_w = (const float*)d_in[15];
  const float* tc_out_b = (const float*)d_in[16];
  const float* emb1 = (const float*)d_in[17];
  const float* emb2 = (const float*)d_in[18];
  const float* lin1_w = (const float*)d_in[19];
  const float* lin1_b = (const float*)d_in[20];
  const float* lin2_w = (const float*)d_in[21];
  const float* lin2_b = (const float*)d_in[22];
  const float* alpha1 = (const float*)d_in[23];
  const float* alpha2 = (const float*)d_in[24];
  const float* alpha3 = (const float*)d_in[25];
  const float* alpha4 = (const float*)d_in[26];
  const float* aff_w = (const float*)d_in[27];
  const float* aff_b = (const float*)d_in[28];
  const float* end0_w = (const float*)d_in[29];
  const float* end0_b = (const float*)d_in[30];
  const float* end1_w = (const float*)d_in[31];
  const float* end1_b = (const float*)d_in[32];
  const float* end2_w = (const float*)d_in[33];
  const float* end2_b = (const float*)d_in[34];

  // ---- workspace carve (~155 MB) ----
  char* p = (char*)d_ws;
  float* s0 = (float*)carve(p, SZ * 4);
  float* s1 = (float*)carve(p, SZ * 4);
  float* s2 = (float*)carve(p, SZ * 4);
  float* dd0 = (float*)carve(p, SZ * 4);
  float* dd1 = (float*)carve(p, SZ * 4);
  float* dd2 = (float*)carve(p, SZ * 4);
  float* t0 = (float*)carve(p, SZ * 4);
  float* t1 = (float*)carve(p, SZ * 4);
  float* t2 = (float*)carve(p, SZ * 4);
  __bf16* zt0 = (__bf16*)carve(p, SZ * 2);
  __bf16* zt1 = (__bf16*)carve(p, SZ * 2);
  __bf16* zt2 = (__bf16*)carve(p, SZ * 2);
  __bf16* ahatA1 = (__bf16*)carve(p, (size_t)4 * NN * 2);
  __bf16* ahatA2 = (__bf16*)carve(p, (size_t)NN * 2);
  __bf16* ahat3  = (__bf16*)carve(p, (size_t)NN * 2);
  __bf16* ahat3T = (__bf16*)carve(p, (size_t)NN * 2);
  float* araw  = (float*)carve(p, (size_t)NN * 4);
  float* adp3m = (float*)carve(p, (size_t)NN * 4);
  float* n1buf = (float*)carve(p, 40960 * 4);
  float* n2buf = (float*)carve(p, 40960 * 4);
  __bf16* wt_in  = (__bf16*)carve(p, 64 * 32 * 2);
  __bf16* wt_hb  = (__bf16*)carve(p, 3 * 64 * 64 * 2);
  __bf16* wt_out = (__bf16*)carve(p, 16 * 64 * 2);
  float* stats = (float*)carve(p, 64);

  const int EB = SZ / 256;                 // elementwise grid = 12288
  const dim3 gg(Fdim / 128, Ndim / 128);   // isip GEMM grid (24, 8)

  // ---- init + graph construction + adjacency prep ----
  k_start<<<EB, 256, 0, stream>>>(input, start_w, start_b, s0, s1, s2);
  k_embmlp<<<160, 256, 0, stream>>>(emb1, lin1_w, lin1_b, n1buf);
  k_embmlp<<<160, 256, 0, stream>>>(emb2, lin2_w, lin2_b, n2buf);
  k_adjraw<<<4096, 256, 0, stream>>>(n1buf, n2buf, araw);
  k_topk<<<1024, 256, 0, stream>>>(araw, adp3m);
  for (int o = 0; o < 4; ++o)
    k_rownorm<<<1024, 256, 0, stream>>>(adjA1 + (size_t)o * NN,
                                        ahatA1 + (size_t)o * NN, 0);
  k_rownorm<<<1024, 256, 0, stream>>>(adjA2, ahatA2, 0);
  k_rownorm<<<1024, 256, 0, stream>>>(adp3m, ahat3, 0);
  k_rownorm<<<1024, 256, 0, stream>>>(adp3m, ahat3T, 1);
  k_prep_tcw<<<1, 256, 0, stream>>>(tc_in_w, tc_h_w, tc_out_w,
                                    wt_in, wt_hb, wt_out);

  // ---- outer Euler loop (4 f-evals) ----
  for (int order = 0; order < 4; ++order) {
    const __bf16* A1 = ahatA1 + (size_t)order * NN;
    // x0 path: relu(conv(cat(x2,x0))) -> tconv -> isip(adjA1[order]) x2
    k_conv_cat_relu<<<EB, 256, 0, stream>>>(s2, s0, mlp1_w, mlp1_b, t0, nullptr);
    k_tconv<<<2048, 256, 0, stream>>>(t0, t1, zt0, wt_in, wt_hb, wt_out,
                                      tc_in_b, tc_h_b, tc_out_b);
    k_isip<<<gg, 256, 0, stream>>>(A1, t1, zt0, t2, zt1, alpha1, 0.5f);
    k_isip<<<gg, 256, 0, stream>>>(A1, t2, zt1, dd0, zt0, alpha1, 0.5f);
    // x1 path: relu(conv(cat(x0_new,x1))) -> isip(adjA2) x2
    k_conv_cat_relu<<<EB, 256, 0, stream>>>(dd0, s1, mlp2_w, mlp2_b, t0, zt0);
    k_isip<<<gg, 256, 0, stream>>>(ahatA2, t0, zt0, t1, zt1, alpha2, 0.5f);
    k_isip<<<gg, 256, 0, stream>>>(ahatA2, t1, zt1, dd1, zt2, alpha2, 0.5f);
    // x2 path: relu(conv(cat(x1_new,x2))) -> isip(adp3)+isip(adp3^T)
    k_conv_cat_relu<<<EB, 256, 0, stream>>>(dd1, s2, mlp3_w, mlp3_b, t0, zt0);
    k_isip<<<gg, 256, 0, stream>>>(ahat3, t0, zt0, t1, zt1, alpha3, 0.5f);
    k_isip<<<gg, 256, 0, stream>>>(ahat3, t1, zt1, t2, zt2, alpha3, 0.5f);
    k_isip<<<gg, 256, 0, stream>>>(ahat3T, t0, zt0, t1, zt1, alpha4, 0.5f);
    k_isip<<<gg, 256, 0, stream>>>(ahat3T, t1, zt1, dd2, zt2, alpha4, 0.5f);
    k_addinto<<<EB, 256, 0, stream>>>(dd2, t2);
    // state += 0.25 * d
    k_axpy<<<EB, 256, 0, stream>>>(s0, dd0);
    k_axpy<<<EB, 256, 0, stream>>>(s1, dd1);
    k_axpy<<<EB, 256, 0, stream>>>(s2, dd2);
  }

  // ---- layernorm + affine + head ----
  hipMemsetAsync(stats, 0, 64, stream);
  k_lnstat<<<dim3(1024, 4), 256, 0, stream>>>(s2, stats);
  k_lnfin<<<1, 32, 0, stream>>>(stats);
  k_head0<<<EB, 256, 0, stream>>>(s2, stats, aff_w, aff_b, end0_w, end0_b, t0);
  k_head1<<<EB / 2, 256, 0, stream>>>(t0, end1_w, end1_b, t1);
  k_head2<<<192, 256, 0, stream>>>(t1, end2_w, end2_b, (float*)d_out);
}
